// FusedMLP5SoftplusWN_69286412419484
// MI455X (gfx1250) — matrix operand order repaired and checked by reading, not compiled, not run on hardware
//
#include <hip/hip_runtime.h>
#include <hip/hip_bf16.h>
#include <math.h>

// ---------------------------------------------------------------------------
// Fused 5-layer weight-normalized MLP with softplus, for MI455X (gfx1250).
//   x[65536,512] -> 1024 -> 1024 -> 1024 -> 1024 -> 512
// bf16 V_WMMA_F32_16X16X32_BF16 with fp32 accumulation; all five layers
// fused per 64-row tile; activations ping-ponged in 256KB LDS (CDNA5 WGP
// has 320KB); weights pre-normalized + B-swizzled to bf16, L2-resident.
// ---------------------------------------------------------------------------

typedef __attribute__((ext_vector_type(16))) __bf16 v16bf;
typedef __attribute__((ext_vector_type(8)))  float  v8f;

#define M_TOTAL 65536
#define K0      512
#define HDIM    1024
#define NOUT    512
#define ROWS    64          // rows per workgroup
#define THREADS 256         // 8 waves (wave32)
#define WS_WEIGHT_BYTE_OFF 32768

__device__ __forceinline__ float softplus_f(float x) {
  // branchless numerically-stable softplus: max(x,0) + log1p(exp(-|x|))
  return fmaxf(x, 0.f) + log1pf(__expf(-fabsf(x)));
}

// ---------------------------------------------------------------------------
// Prep 1: per-column weight-norm scale  s[j] = g[j] / ||v[:,j]||
// One thread per column; reads are coalesced across threads (stride-N rows).
// ---------------------------------------------------------------------------
__global__ void wn_scale_kernel(const float* __restrict__ v,
                                const float* __restrict__ g,
                                float* __restrict__ scale, int K, int N) {
  int j = blockIdx.x * blockDim.x + threadIdx.x;
  if (j >= N) return;
  float s = 0.f;
  for (int k = 0; k < K; ++k) {
    float t = v[(size_t)k * N + j];
    s += t * t;
  }
  scale[j] = g[j] * rsqrtf(s);
}

// ---------------------------------------------------------------------------
// Prep 2: write normalized weight as bf16 in WMMA-B tile-packed layout.
// B tile is 32x16 (KxN): lane = ((k>>3)&1)*16 + (j&15), element = (k&7) | (((k>>4)&1)<<3).
// Tile (kt,nt) occupies 512 bf16 = 1KB contiguous: lane*16 + e.
// ---------------------------------------------------------------------------
__global__ void wn_swizzle_kernel(const float* __restrict__ v,
                                  const float* __restrict__ scale,
                                  __bf16* __restrict__ wout, int K, int N) {
  int idx = blockIdx.x * blockDim.x + threadIdx.x;
  if (idx >= K * N) return;
  int k = idx / N, j = idx - k * N;
  float w = v[idx] * scale[j];
  int kt = k >> 5, nt = j >> 4;
  int lane = (((k >> 3) & 1) << 4) | (j & 15);
  int e = (k & 7) | (((k >> 4) & 1) << 3);
  int Ntiles = N >> 4;
  wout[((size_t)(kt * Ntiles + nt) << 9) + (lane << 4) + e] = (__bf16)w;
}

// ---------------------------------------------------------------------------
// A-layout (16-bit A matrix 16x32 per ISA 7.12.2):
//   within a 32-wide K tile: gsel = (kk>>3)&1 ; e = (kk&7) | ((kk>>4)<<3)
//   lane = gsel*16 + (m&15); packed tile = 512 bf16, offset lane*16 + e.
// ---------------------------------------------------------------------------

__global__ __launch_bounds__(THREADS, 1)
void fused_mlp5_wmma(const float* __restrict__ x,
                     const __bf16* __restrict__ wbase,
                     float* __restrict__ out) {
  // 2 x (64 rows x 1024 K) bf16 = 2 x 128KB ping-pong activation buffers
  __shared__ __align__(16) __bf16 smem[2][ROWS * HDIM];

  const int tid   = threadIdx.x;
  const int lane  = tid & 31;
  const int wave  = tid >> 5;
  const int wgRow = blockIdx.x * ROWS;

  // ---- stage x (fp32 -> bf16) into smem[0] in A-tile layout ----
  {
    const int Kt0 = K0 >> 5; // 16 K-tiles
    for (int idx = tid; idx < ROWS * K0; idx += THREADS) {
      int m = idx >> 9;            // row within tile block (K0 = 512)
      int k = idx & (K0 - 1);
      float val = x[(size_t)(wgRow + m) * K0 + k];
      int kt = k >> 5, kk = k & 31;
      int gsel = (kk >> 3) & 1;
      int e    = (kk & 7) | ((kk >> 4) << 3);
      int l2   = (gsel << 4) | (m & 15);
      int mt   = m >> 4;
      smem[0][((mt * Kt0 + kt) << 9) + (l2 << 4) + e] = (__bf16)val;
    }
  }
  __syncthreads();

  const int    Kdims[5] = {512, 1024, 1024, 1024, 1024};
  const int    Ndims[5] = {1024, 1024, 1024, 1024, 512};
  const size_t Woff[5]  = {0, 524288, 1572864, 2621440, 3670016};

  const v8f vzero = {0.f, 0.f, 0.f, 0.f, 0.f, 0.f, 0.f, 0.f};

  #pragma unroll
  for (int l = 0; l < 5; ++l) {
    const int Kt      = Kdims[l] >> 5;   // 32-wide K tiles
    const int Ntiles  = Ndims[l] >> 4;   // 16-wide N tiles
    const int npasses = Ndims[l] >> 8;   // 256 output cols per pass (8 waves x 32)
    const int KtNext  = Ndims[l] >> 5;   // next layer's K tiles
    const __bf16* __restrict__ W    = wbase + Woff[l];
    const __bf16* __restrict__ aIn  = smem[l & 1];
    __bf16* __restrict__       aOut = smem[(l + 1) & 1];

    for (int p = 0; p < npasses; ++p) {
      const int nb = (p << 4) + (wave << 1);  // first of this wave's two N tiles

      v8f acc[4][2];
      #pragma unroll
      for (int mt = 0; mt < 4; ++mt) { acc[mt][0] = vzero; acc[mt][1] = vzero; }

      for (int kt = 0; kt < Kt; ++kt) {
        const __bf16* wp = W + ((size_t)(kt * Ntiles + nb) << 9) + (lane << 4);
        v16bf b0 = *(const v16bf*)(wp);
        v16bf b1 = *(const v16bf*)(wp + 512);
        if (kt + 1 < Kt)  // WGP-scope prefetch of next K-slab (fills all levels)
          __builtin_prefetch(wp + ((size_t)Ntiles << 9), 0, 3);
        #pragma unroll
        for (int mt = 0; mt < 4; ++mt) {
          v16bf a = *(const v16bf*)(aIn + ((mt * Kt + kt) << 9) + (lane << 4));
          acc[mt][0] = __builtin_amdgcn_wmma_f32_16x16x32_bf16(
              false, a, false, b0, (short)0, acc[mt][0], false, false);
          acc[mt][1] = __builtin_amdgcn_wmma_f32_16x16x32_bf16(
              false, a, false, b1, (short)0, acc[mt][1], false, false);
        }
      }

      // ---- epilogue ----
      const int mhalf = (lane >> 4) << 3;  // C layout: M = r + 8*(lane>=16)
      const int ncol  = lane & 15;
      if (l < 4) {
        // softplus -> bf16 -> next layer's A layout in LDS
        #pragma unroll
        for (int mt = 0; mt < 4; ++mt) {
          #pragma unroll
          for (int t = 0; t < 2; ++t) {
            const int nt_abs = nb + t;
            const int kk   = ((nt_abs & 1) << 4) | ncol; // K index within next 32-tile
            const int ktn  = nt_abs >> 1;
            const int gsel = (kk >> 3) & 1;
            const int e    = (kk & 7) | ((kk >> 4) << 3);
            #pragma unroll
            for (int r = 0; r < 8; ++r) {
              float val = softplus_f(acc[mt][t][r]);
              int l2 = (gsel << 4) | (mhalf + r);
              aOut[((mt * KtNext + ktn) << 9) + (l2 << 4) + e] = (__bf16)val;
            }
          }
        }
      } else {
        // final linear layer -> fp32 global store
        #pragma unroll
        for (int mt = 0; mt < 4; ++mt) {
          #pragma unroll
          for (int t = 0; t < 2; ++t) {
            const int col = ((nb + t) << 4) | ncol;
            #pragma unroll
            for (int r = 0; r < 8; ++r) {
              int row = wgRow + (mt << 4) + mhalf + r;
              out[(size_t)row * NOUT + col] = acc[mt][t][r];
            }
          }
        }
      }
    }
    __syncthreads();  // all waves done with aIn / aOut before ping-pong
  }
}

// ---------------------------------------------------------------------------
// Host launcher
// ---------------------------------------------------------------------------
extern "C" void kernel_launch(void* const* d_in, const int* in_sizes, int n_in,
                              void* d_out, int out_size, void* d_ws, size_t ws_size,
                              hipStream_t stream) {
  (void)in_sizes; (void)n_in; (void)out_size; (void)ws_size;

  const float* x = (const float*)d_in[0];
  const float* v[5] = {(const float*)d_in[1], (const float*)d_in[3],
                       (const float*)d_in[5], (const float*)d_in[7],
                       (const float*)d_in[9]};
  const float* g[5] = {(const float*)d_in[2], (const float*)d_in[4],
                       (const float*)d_in[6], (const float*)d_in[8],
                       (const float*)d_in[10]};
  float* out = (float*)d_out;

  float*  scales = (float*)d_ws;
  __bf16* wbase  = (__bf16*)((char*)d_ws + WS_WEIGHT_BYTE_OFF);

  const int    Kd[5]   = {512, 1024, 1024, 1024, 1024};
  const int    Nd[5]   = {1024, 1024, 1024, 1024, 512};
  const size_t Woff[5] = {0, 524288, 1572864, 2621440, 3670016};

  size_t soff = 0;
  for (int l = 0; l < 5; ++l) {
    wn_scale_kernel<<<(Nd[l] + 255) / 256, 256, 0, stream>>>(
        v[l], g[l], scales + soff, Kd[l], Nd[l]);
    wn_swizzle_kernel<<<(Kd[l] * Nd[l]) / 256, 256, 0, stream>>>(
        v[l], scales + soff, wbase + Woff[l], Kd[l], Nd[l]);
    soff += Nd[l];
  }

  fused_mlp5_wmma<<<M_TOTAL / ROWS, THREADS, 0, stream>>>(x, wbase, out);
}
